// Sinkhorn_75900662055161
// MI455X (gfx1250) — matrix operand order repaired and checked by reading, not compiled, not run on hardware
//
#include <hip/hip_runtime.h>

// ---------------------------------------------------------------------------
// Sinkhorn divergence (geomloss defaults: p=2, blur=0.5, scaling=0.9, debias)
// for x,y in R^{4096 x 32}, fully fused on gfx1250 (MI455X).
//
// Core kernel: fused softmin. For rows R and columns C with duals g:
//   out_i = 0.5*|r_i|^2 - eps*( wlog + ln2*( M_i + log2 sum_j 2^(t_ij - M_i) ) )
//   t_ij  = ( (g_j - 0.5*|c_j|^2) + r_i . c_j ) * (log2e/eps)
// dot products via v_wmma_f32_16x16x32_f16 (K=32 == D, one WMMA per tile).
// Two-pass (max, then sum) log2-domain softmax => 1 v_exp_f32 per element.
// ---------------------------------------------------------------------------

typedef __attribute__((ext_vector_type(16))) _Float16 v16h;
typedef __attribute__((ext_vector_type(8)))  float    v8f;

#define NPTS 4096
#define DIM  32
#define NMAX_ITERS 56           // >= device-computed schedule length L (~45)
#define TPB 128                 // 4 waves of 32
#define WLOG (-8.31776616671934f)   // -ln(4096) uniform log-weights
#define LOG2E 1.44269504088896340f
#define LN2   0.69314718055994531f

// ---------------------------------------------------------------------------
// Prep: squared norms, f32->f16 conversion, per-block per-dim min/max keys
// (order-preserving uint map so integer atomics give float min/max).
// grid: 32 blocks x 256 threads over concat(x,y) = 8192 points.
// ---------------------------------------------------------------------------
__global__ void prep_kernel(const float* __restrict__ x, const float* __restrict__ y,
                            float* __restrict__ sq, _Float16* __restrict__ h16,
                            unsigned* __restrict__ pmin, unsigned* __restrict__ pmax) {
    __shared__ unsigned smin[DIM], smax[DIM];
    int t = threadIdx.x;
    if (t < DIM) { smin[t] = 0xFFFFFFFFu; smax[t] = 0u; }
    __syncthreads();

    int g = blockIdx.x * 256 + t;                      // 0..8191
    const float* p = (g < NPTS) ? (x + (size_t)g * DIM)
                                : (y + (size_t)(g - NPTS) * DIM);
    float ss = 0.0f;
#pragma unroll
    for (int d = 0; d < DIM; ++d) {
        float v = p[d];
        ss += v * v;
        h16[(size_t)g * DIM + d] = (_Float16)v;
        unsigned u = __float_as_uint(v);
        unsigned key = (u & 0x80000000u) ? ~u : (u | 0x80000000u);
        atomicMin(&smin[d], key);
        atomicMax(&smax[d], key);
    }
    sq[g] = ss;
    __syncthreads();
    if (t < DIM) {
        pmin[blockIdx.x * DIM + t] = smin[t];
        pmax[blockIdx.x * DIM + t] = smax[t];
    }
}

// ---------------------------------------------------------------------------
// Finalize: reduce partials -> diameter -> eps schedule (numpy-faithful, f64).
// 1 block x 32 threads (one per dim).
// ---------------------------------------------------------------------------
__global__ void finalize_kernel(const unsigned* __restrict__ pmin,
                                const unsigned* __restrict__ pmax,
                                float* __restrict__ eps, int* __restrict__ Lout) {
    int d = threadIdx.x;                               // 0..31
    unsigned kmin = 0xFFFFFFFFu, kmax = 0u;
    for (int b = 0; b < 32; ++b) {
        unsigned a = pmin[b * DIM + d]; if (a < kmin) kmin = a;
        unsigned c = pmax[b * DIM + d]; if (c > kmax) kmax = c;
    }
    float lo = (kmin & 0x80000000u) ? __uint_as_float(kmin & 0x7FFFFFFFu)
                                    : __uint_as_float(~kmin);
    float hi = (kmax & 0x80000000u) ? __uint_as_float(kmax & 0x7FFFFFFFu)
                                    : __uint_as_float(~kmax);
    float r  = hi - lo;
    float r2 = r * r;
#pragma unroll
    for (int m = 16; m > 0; m >>= 1) r2 += __shfl_xor(r2, m, 32);

    if (d == 0) {
        double diam = sqrt((double)r2);
        double a    = 2.0 * log(diam);                 // P*log(diameter)
        double stop = 2.0 * log(0.5);                  // P*log(blur)
        double step = 2.0 * log(0.9);                  // P*log(scaling)
        int cnt = (int)ceil((stop - a) / step);        // np.arange length
        if (cnt < 0)  cnt = 0;
        if (cnt > 60) cnt = 60;
        eps[0] = (float)(diam * diam);
        for (int k = 0; k < cnt; ++k) eps[1 + k] = (float)exp(a + step * (double)k);
        eps[cnt + 1] = 0.25f;                          // blur^p
        int L = cnt + 2;
        if (L > NMAX_ITERS) { L = NMAX_ITERS; eps[L - 1] = 0.25f; }
        Lout[0] = L;
    }
}

// ---------------------------------------------------------------------------
// Fused 4-way softmin. gridDim.y selects problem:
//   0: f_ba  rows=x cols=y dual=g_ab   1: g_ab rows=y cols=x dual=f_ba
//   2: f_aa  rows=x cols=x dual=f_aa   3: g_bb rows=y cols=y dual=g_bb
// mode: 0=init (dual=0, direct write), 1=iter (averaged, guarded by L),
//       2=final extrapolation (direct write, eps = eps_list[L-1]).
// Block: 128 threads = 4 waves; each wave owns one 16-row tile; loops 256
// column tiles; WMMA f16 K=32 per tile; two-pass log2-domain softmax.
// ---------------------------------------------------------------------------
__global__ void softmin4_kernel(const _Float16* __restrict__ H16,   // x16 | y16
                                const float* __restrict__ SQ,       // x2  | y2
                                const float* __restrict__ srcSet,   // 4 x NPTS
                                float* __restrict__ dstSet,         // 4 x NPTS
                                const float* __restrict__ epsList,
                                const int* __restrict__ Lptr,
                                int mode, int iter) {
    __shared__ float sh_hy[NPTS];                      // 16 KB

    const int p      = blockIdx.y;
    const int L      = Lptr[0];
    const int rowsX  = (p == 0 || p == 2);
    const int colsX  = (p == 1 || p == 2);
    const _Float16* R16 = H16 + (rowsX ? 0 : (size_t)NPTS * DIM);
    const _Float16* C16 = H16 + (colsX ? 0 : (size_t)NPTS * DIM);
    const float*    rsq = SQ + (rowsX ? 0 : NPTS);
    const float*    csq = SQ + (colsX ? 0 : NPTS);
    const int dualIdx   = (p == 0) ? 1 : (p == 1) ? 0 : p;
    const float* dual   = srcSet + dualIdx * NPTS;
    const float* oldv   = srcSet + p * NPTS;
    float*       out    = dstSet + p * NPTS;

    const int t = threadIdx.x;

    if (mode == 1 && iter >= L) {                      // schedule exhausted: copy
        int rb = blockIdx.x * 64;
        for (int j = t; j < 64; j += TPB) out[rb + j] = oldv[rb + j];
        return;
    }

    const int   epsIdx = (mode == 0) ? 0 : (mode == 2) ? (L - 1) : iter;
    const float eps    = epsList[epsIdx];
    const float ie2    = (1.0f / eps) * LOG2E;         // log2e / eps

    // hy_j = (g_j - 0.5*|c_j|^2) * log2e/eps   (g=0 for init); wlog pulled out.
    for (int j = t; j < NPTS; j += TPB) {
        float g = (mode == 0) ? 0.0f : dual[j];
        sh_hy[j] = (g - 0.5f * csq[j]) * ie2;
    }
    __syncthreads();

    const int lane = t & 31;
    const int r16  = lane & 15;
    const int hhalf = lane >> 4;
    const int rowBase = blockIdx.x * 64 + (t >> 5) * 16;

    // A tile (loop-invariant): lane<16 -> K 0..7 & 16..23 ; lane>=16 -> K 8..15 & 24..31
    union AV { v16h v; uint4 q[2]; };
    const _Float16* arow = R16 + (size_t)(rowBase + r16) * DIM + hhalf * 8;
    AV A;
    A.q[0] = *(const uint4*)(arow);
    A.q[1] = *(const uint4*)(arow + 16);

    // ---- pass 1: per-row maxima ----
    float M[8];
#pragma unroll
    for (int r = 0; r < 8; ++r) M[r] = -3.0e38f;

    for (int ct = 0; ct < NPTS / 16; ++ct) {
        const _Float16* brow = C16 + (size_t)(ct * 16 + r16) * DIM + hhalf * 8;
        AV B;
        B.q[0] = *(const uint4*)(brow);
        B.q[1] = *(const uint4*)(brow + 16);
        v8f c = {};
        c = __builtin_amdgcn_wmma_f32_16x16x32_f16(false, A.v, false, B.v,
                                                   (short)0, c, false, false);
        float hy = sh_hy[ct * 16 + r16];
#pragma unroll
        for (int r = 0; r < 8; ++r)
            M[r] = fmaxf(M[r], fmaf(c[r], ie2, hy));
    }
#pragma unroll
    for (int m = 1; m < 16; m <<= 1)
#pragma unroll
        for (int r = 0; r < 8; ++r)
            M[r] = fmaxf(M[r], __shfl_xor(M[r], m, 32));

    // ---- pass 2: sum 2^(t - M) with shared per-row M ----
    float S[8];
#pragma unroll
    for (int r = 0; r < 8; ++r) S[r] = 0.0f;

    for (int ct = 0; ct < NPTS / 16; ++ct) {
        const _Float16* brow = C16 + (size_t)(ct * 16 + r16) * DIM + hhalf * 8;
        AV B;
        B.q[0] = *(const uint4*)(brow);
        B.q[1] = *(const uint4*)(brow + 16);
        v8f c = {};
        c = __builtin_amdgcn_wmma_f32_16x16x32_f16(false, A.v, false, B.v,
                                                   (short)0, c, false, false);
        float hy = sh_hy[ct * 16 + r16];
#pragma unroll
        for (int r = 0; r < 8; ++r)
            S[r] += __builtin_amdgcn_exp2f(fmaf(c[r], ie2, hy) - M[r]);
    }
#pragma unroll
    for (int m = 1; m < 16; m <<= 1)
#pragma unroll
        for (int r = 0; r < 8; ++r)
            S[r] += __shfl_xor(S[r], m, 32);

    // ---- epilogue: lanes with (lane&15)<8 write rows ----
    if (r16 < 8) {
        int r   = r16;
        int row = rowBase + r + 8 * hhalf;
        float lse = (M[r] + __builtin_amdgcn_logf(S[r])) * LN2;   // natural-log LSE
        float val = 0.5f * rsq[row] - eps * (WLOG + lse);
        if (mode == 1) val = 0.5f * (oldv[row] + val);
        out[row] = val;
    }
}

// ---------------------------------------------------------------------------
// Final reduction: mean(f_ba - f_aa) + mean(g_ab - g_bb)
// ---------------------------------------------------------------------------
__global__ void reduce_kernel(const float* __restrict__ fin, float* __restrict__ out) {
    __shared__ float sh[256];
    float acc = 0.0f;
    for (int j = threadIdx.x; j < NPTS; j += 256)
        acc += (fin[j] - fin[2 * NPTS + j]) + (fin[NPTS + j] - fin[3 * NPTS + j]);
    sh[threadIdx.x] = acc;
    __syncthreads();
    for (int s = 128; s > 0; s >>= 1) {
        if (threadIdx.x < s) sh[threadIdx.x] += sh[threadIdx.x + s];
        __syncthreads();
    }
    if (threadIdx.x == 0) out[0] = sh[0] / 4096.0f;
}

// ---------------------------------------------------------------------------
// Host side: fixed launch sequence (graph-capture safe, deterministic).
// Workspace layout (float index into W):
//   Wi[0]      : L (int)
//   W[16..79]  : eps_list
//   W[128..]   : pmin keys (1024), W[1152..]: pmax keys (1024)
//   W[4096..]  : sq = x2(4096) | y2(4096)
//   W[12288..] : set0(4x4096) | set1(4x4096) | final(4x4096)
//   W[65536..] : f16 points  x16(4096x32) | y16(4096x32)   (512 KB)
// Total ~768 KB of d_ws.
// ---------------------------------------------------------------------------
extern "C" void kernel_launch(void* const* d_in, const int* in_sizes, int n_in,
                              void* d_out, int out_size, void* d_ws, size_t ws_size,
                              hipStream_t stream) {
    const float* x = (const float*)d_in[0];
    const float* y = (const float*)d_in[1];

    float*     W    = (float*)d_ws;
    int*       Wi   = (int*)d_ws;
    float*     eps  = W + 16;
    unsigned*  pmin = (unsigned*)(W + 128);
    unsigned*  pmax = (unsigned*)(W + 1152);
    float*     sq   = W + 4096;
    float*     set0 = W + 12288;
    float*     set1 = set0 + 4 * NPTS;
    float*     setF = set1 + 4 * NPTS;
    _Float16*  h16  = (_Float16*)(W + 65536);

    prep_kernel<<<32, 256, 0, stream>>>(x, y, sq, h16, pmin, pmax);
    finalize_kernel<<<1, 32, 0, stream>>>(pmin, pmax, eps, Wi);

    // init at eps_list[0] (direct write into set0; src unused in mode 0)
    softmin4_kernel<<<dim3(64, 4), TPB, 0, stream>>>(h16, sq, set0, set0, eps, Wi, 0, 0);

    float* bufs[2] = {set0, set1};
    for (int i = 0; i < NMAX_ITERS; ++i)
        softmin4_kernel<<<dim3(64, 4), TPB, 0, stream>>>(h16, sq, bufs[i & 1],
                                                         bufs[(i + 1) & 1], eps, Wi, 1, i);

    // final extrapolation at eps = eps_list[L-1]
    softmin4_kernel<<<dim3(64, 4), TPB, 0, stream>>>(h16, sq, bufs[NMAX_ITERS & 1],
                                                     setF, eps, Wi, 2, 0);

    reduce_kernel<<<1, 256, 0, stream>>>(setF, (float*)d_out);
}
